// GraphAttentionLayer_83416854823613
// MI455X (gfx1250) — compile-verified, matrix operand order checked by reading, not run on hardware
//
#include <hip/hip_runtime.h>
#include <hip/hip_bf16.h>

typedef __attribute__((ext_vector_type(16))) __bf16 v16bf;
typedef __attribute__((ext_vector_type(8)))  float  v8f;

#define IN_F  256
#define NJ    256   // HEADS*OUT_F
#define ALPHA 0.2f

// ---------------------------------------------------------------------------
// Kernel 1: h[n, j] = sum_k x[n,k] * W2d[j,k], W2d = W viewed as [256,256]
// (row j = head*32 + f). One wave computes one 16x16 tile of h via
// v_wmma_f32_16x16x32_bf16 with bf16 hi/lo split for ~fp32 accuracy.
// ---------------------------------------------------------------------------
__global__ __launch_bounds__(32)
void gat_gemm_wmma(const float* __restrict__ x,
                   const float* __restrict__ W,   // [NJ, IN_F] row-major
                   float* __restrict__ h,         // [N, NJ]
                   int N) {
    const int mtile = blockIdx.x;         // row tile
    const int ctile = blockIdx.y;         // col tile (0..15)
    const int lane  = threadIdx.x;        // 0..31
    const int half  = lane >> 4;          // 0 or 1
    const int l16   = lane & 15;

    int row = mtile * 16 + l16;
    int arow = (row < N) ? row : (N - 1);           // clamp (N%16==0 here)
    const float* __restrict__ ar = x + (size_t)arow * IN_F;
    const float* __restrict__ br = W + (size_t)(ctile * 16 + l16) * IN_F;

    v8f acc = {};
    for (int k0 = 0; k0 < IN_F; k0 += 32) {
        // Per-lane K map (ISA 16-bit A layout): lanes 0-15 -> K = k0+0..7,16..23
        //                                       lanes 16-31-> K = k0+8..15,24..31
        const int kb = k0 + half * 8;
        v16bf ahi, alo, bhi, blo;
#pragma unroll
        for (int i = 0; i < 8; ++i) {
            float a0 = ar[kb + i];
            float a1 = ar[kb + 16 + i];
            float b0 = br[kb + i];
            float b1 = br[kb + 16 + i];
            __bf16 t;
            t = (__bf16)a0; ahi[i]     = t; alo[i]     = (__bf16)(a0 - (float)t);
            t = (__bf16)a1; ahi[8 + i] = t; alo[8 + i] = (__bf16)(a1 - (float)t);
            t = (__bf16)b0; bhi[i]     = t; blo[i]     = (__bf16)(b0 - (float)t);
            t = (__bf16)b1; bhi[8 + i] = t; blo[8 + i] = (__bf16)(b1 - (float)t);
        }
        // acc += hi*hi + hi*lo + lo*hi  (bf16x3 split ~= fp32 GEMM)
        acc = __builtin_amdgcn_wmma_f32_16x16x32_bf16(false, ahi, false, bhi,
                                                      (short)0, acc, false, false);
        acc = __builtin_amdgcn_wmma_f32_16x16x32_bf16(false, ahi, false, blo,
                                                      (short)0, acc, false, false);
        acc = __builtin_amdgcn_wmma_f32_16x16x32_bf16(false, alo, false, bhi,
                                                      (short)0, acc, false, false);
    }

    // D layout: VGPR r -> M = 8*half + r, N = l16.
    // Wave-uniform bounds check: whole 16-row tile in range -> branch-free stores.
    float* __restrict__ hp = h + ((size_t)(mtile * 16 + half * 8)) * NJ
                               + ctile * 16 + l16;
    if (mtile * 16 + 16 <= N) {
#pragma unroll
        for (int r = 0; r < 8; ++r) {
            hp[(size_t)r * NJ] = acc[r];
        }
    } else {
#pragma unroll
        for (int r = 0; r < 8; ++r) {
            int m = mtile * 16 + half * 8 + r;
            if (m < N) hp[(size_t)r * NJ] = acc[r];
        }
    }
}

// ---------------------------------------------------------------------------
// Kernel 2: per-edge attention + scatter. One wave per edge.
// Lane L owns columns [L*8, L*8+8) (all in head L/4).
// ---------------------------------------------------------------------------
__global__ __launch_bounds__(256)
void gat_edge(const float* __restrict__ h,
              const int* __restrict__ src,
              const int* __restrict__ dst,
              float* __restrict__ hprime,    // [N, NJ] accumulators (zeroed)
              float* __restrict__ rowsum,    // [N, 8]  accumulators (zeroed)
              int E) {
    const int lane = threadIdx.x & 31;
    const int wib  = threadIdx.x >> 5;
    const int e = blockIdx.x * 8 + wib;
    if (e >= E) return;

    const int s = src[e];
    const int d = dst[e];

    const float4* __restrict__ hs4 =
        (const float4*)(h + (size_t)s * NJ) + lane * 2;
    const float4* __restrict__ hd4 =
        (const float4*)(h + (size_t)d * NJ) + lane * 2;
    float4 a0 = hs4[0], a1 = hs4[1];
    float4 b0 = hd4[0], b1 = hd4[1];

    float as[8] = {a0.x, a0.y, a0.z, a0.w, a1.x, a1.y, a1.z, a1.w};
    float bs[8] = {b0.x, b0.y, b0.z, b0.w, b1.x, b1.y, b1.z, b1.w};

    float p = 0.0f;
#pragma unroll
    for (int i = 0; i < 8; ++i) {
        float df = as[i] - bs[i];
        p = fmaf(df, df, p);
    }
    // reduce partials within each 4-lane head group (wave32 shuffles)
    p += __shfl_xor(p, 1, 32);
    p += __shfl_xor(p, 2, 32);
    const float sqd = p;                       // per-head sq distance
    const float eh = -sqd;
    const float lr = (eh > 0.0f) ? eh : ALPHA * eh;   // leaky_relu
    const float ee = __expf(-lr);                     // edge weight

    float* __restrict__ hp = hprime + (size_t)s * NJ + lane * 8;
#pragma unroll
    for (int i = 0; i < 8; ++i) {
        atomicAdd(&hp[i], ee * bs[i]);
    }
    if ((lane & 3) == 0) {
        atomicAdd(&rowsum[(size_t)s * 8 + (lane >> 2)], ee);
    }
}

// ---------------------------------------------------------------------------
// Kernel 3: out = elu(hprime / rowsum[head])
// ---------------------------------------------------------------------------
__global__ __launch_bounds__(256)
void gat_finalize(const float* __restrict__ hprime,
                  const float* __restrict__ rowsum,
                  float* __restrict__ out,
                  int total) {
    int idx = blockIdx.x * 256 + threadIdx.x;
    if (idx >= total) return;
    int n = idx >> 8;           // NJ == 256
    int j = idx & 255;
    int head = j >> 5;          // OUT_F == 32
    float v = hprime[idx] / rowsum[(size_t)n * 8 + head];
    out[idx] = (v > 0.0f) ? v : (__expf(v) - 1.0f);
}

// ---------------------------------------------------------------------------
extern "C" void kernel_launch(void* const* d_in, const int* in_sizes, int n_in,
                              void* d_out, int out_size, void* d_ws, size_t ws_size,
                              hipStream_t stream) {
    const float* x  = (const float*)d_in[0];          // [N, 256]
    const int*   ei = (const int*)d_in[1];            // [2, E]
    const float* W  = (const float*)d_in[2];          // [8, 32, 256] == [256, 256]

    const int N = in_sizes[0] / IN_F;
    const int E = in_sizes[1] / 2;
    const int* srcIdx = ei;
    const int* dstIdx = ei + E;

    float* h      = (float*)d_ws;                     // N*256 f32
    float* hprime = h + (size_t)N * NJ;               // N*256 f32
    float* rowsum = hprime + (size_t)N * NJ;          // N*8   f32

    // zero the accumulators (hprime and rowsum are contiguous)
    hipMemsetAsync(hprime, 0,
                   ((size_t)N * NJ + (size_t)N * 8) * sizeof(float), stream);

    dim3 gGemm((N + 15) / 16, NJ / 16);
    gat_gemm_wmma<<<gGemm, 32, 0, stream>>>(x, W, h, N);

    gat_edge<<<(E + 7) / 8, 256, 0, stream>>>(h, srcIdx, dstIdx,
                                              hprime, rowsum, E);

    int total = N * NJ;
    gat_finalize<<<(total + 255) / 256, 256, 0, stream>>>(hprime, rowsum,
                                                          (float*)d_out, total);
}
